// SparseMoEBlock_12841952215337
// MI455X (gfx1250) — matrix operand order repaired and checked by reading, not compiled, not run on hardware
//
#include <hip/hip_runtime.h>
#include <hip/hip_bf16.h>

// ---------------- problem constants ----------------
#define T_TOK 16384   // B*S
#define Hd    1024
#define Id    2048
#define NE    16
#define MT    32      // tokens per workgroup tile
#define XAS   1032    // LDS row stride for x tile (f16 elems), padded
#define ALS   136     // LDS row stride for activation tile
#define TILES_MAX 512 // ceil(T_TOK / MT): worst case all tokens on one expert

typedef __attribute__((ext_vector_type(16))) _Float16 v16h;
typedef __attribute__((ext_vector_type(8)))  float    v8f;
typedef __attribute__((ext_vector_type(4)))  unsigned int u32x4;

union ABFrag { v16h v; u32x4 u[2]; };

#define WMMA_F16(a, b, c) \
  __builtin_amdgcn_wmma_f32_16x16x32_f16(false, (a), false, (b), (short)0, (c), false, false)

// A fragment (16x32 f16) from an LDS tile, rows = mbase..mbase+15, K = kk..kk+31.
// Layout (ISA 7.12.2): lane<16: M=lane, halves K {kk..kk+7, kk+16..kk+23};
//                      lane>=16: M=lane-16, halves K {kk+8..kk+15, kk+24..kk+31}.
__device__ inline v16h ldsA(const _Float16* base, int stride, int mbase, int kk, int lane) {
  int lr = lane & 15;
  int hi = lane >> 4;                       // 0 or 1
  const _Float16* p = base + (mbase + lr) * stride + kk + hi * 8;
  ABFrag t;
  t.u[0] = *(const u32x4*)p;
  t.u[1] = *(const u32x4*)(p + 16);
  return t.v;
}

// B fragment (32x16 f16) from global weights stored [N][K] (K contiguous).
// Layout: lane<16: N=lane, K {kk..kk+7, kk+16..kk+23} ; lane>=16: N=lane-16, K {kk+8..kk+15, kk+24..kk+31}.
__device__ inline v16h globB(const _Float16* base, int stride, int nbase, int kk, int lane) {
  int lr = lane & 15;
  int hi = lane >> 4;
  const _Float16* p = base + (size_t)(nbase + lr) * stride + kk + hi * 8;
  ABFrag t;
  t.u[0] = *(const u32x4*)p;
  t.u[1] = *(const u32x4*)(p + 16);
  return t.v;
}

// ---------------- router: softmax -> top-2 -> renorm -> expert lists ----------------
__global__ __launch_bounds__(256) void router_kernel(
    const float* __restrict__ x, const float* __restrict__ wg,
    int* __restrict__ counts, int* __restrict__ tlist, float* __restrict__ wlist) {
  int lane = threadIdx.x & 31;
  int tok = blockIdx.x * 8 + (threadIdx.x >> 5);   // one wave per token
  if (tok >= T_TOK) return;

  float acc[NE];
#pragma unroll
  for (int j = 0; j < NE; ++j) acc[j] = 0.f;

  const float* xr = x + (size_t)tok * Hd;
  for (int h = lane; h < Hd; h += 32) {
    float xv = xr[h];
    const float* wrow = wg + (size_t)h * NE;
#pragma unroll
    for (int j = 0; j < NE; ++j) acc[j] += xv * wrow[j];
  }
#pragma unroll
  for (int off = 16; off > 0; off >>= 1) {
#pragma unroll
    for (int j = 0; j < NE; ++j) acc[j] += __shfl_down(acc[j], off, 32);
  }
  if (lane == 0) {
    float mx = acc[0];
#pragma unroll
    for (int j = 1; j < NE; ++j) mx = fmaxf(mx, acc[j]);
    float p[NE];
#pragma unroll
    for (int j = 0; j < NE; ++j) p[j] = __expf(acc[j] - mx);
    // top-1
    int i1 = 0; float v1 = p[0];
#pragma unroll
    for (int j = 1; j < NE; ++j) if (p[j] > v1) { v1 = p[j]; i1 = j; }
    // top-2
    int i2 = (i1 == 0) ? 1 : 0; float v2 = p[i2];
#pragma unroll
    for (int j = 0; j < NE; ++j) if (j != i1 && p[j] > v2) { v2 = p[j]; i2 = j; }
    float inv = 1.f / (v1 + v2);
    float w1 = v1 * inv, w2 = v2 * inv;
    int pos1 = atomicAdd(&counts[i1], 1);
    tlist[i1 * T_TOK + pos1] = tok; wlist[i1 * T_TOK + pos1] = w1;
    int pos2 = atomicAdd(&counts[i2], 1);
    tlist[i2 * T_TOK + pos2] = tok; wlist[i2 * T_TOK + pos2] = w2;
  }
}

// ---------------- fp32 -> f16 activation convert ----------------
__global__ __launch_bounds__(256) void cvt_x_kernel(
    const float* __restrict__ in, _Float16* __restrict__ out, long n8) {
  long i = (long)blockIdx.x * 256 + threadIdx.x;
  if (i >= n8) return;
  const float* p = in + i * 8;
  _Float16 h[8];
#pragma unroll
  for (int j = 0; j < 8; ++j) h[j] = (_Float16)p[j];
  *(u32x4*)(out + i * 8) = *(const u32x4*)h;
}

// ---------------- batched transpose + convert: f32 [R][C] -> f16 [C][R] ----------------
__global__ __launch_bounds__(256) void transpose_cvt_kernel(
    const float* __restrict__ in, _Float16* __restrict__ out, int R, int C) {
  __shared__ float tile[32][33];
  size_t base = (size_t)blockIdx.z * R * C;
  int c0 = blockIdx.x * 32, r0 = blockIdx.y * 32;
#pragma unroll
  for (int i = threadIdx.y; i < 32; i += 8)
    tile[i][threadIdx.x] = in[base + (size_t)(r0 + i) * C + c0 + threadIdx.x];
  __syncthreads();
#pragma unroll
  for (int i = threadIdx.y; i < 32; i += 8)
    out[base + (size_t)(c0 + i) * R + r0 + threadIdx.x] = (_Float16)tile[threadIdx.x][i];
}

// ---------------- fused expert FFN: (x@Wg, x@Wu) -> silu -> @Wd -> atomic combine ----------------
__global__ __launch_bounds__(256) void moe_ffn_kernel(
    const _Float16* __restrict__ xh,
    const _Float16* __restrict__ wg_h,   // [E][I][H]  (N-major, K contiguous)
    const _Float16* __restrict__ wu_h,   // [E][I][H]
    const _Float16* __restrict__ wd_h,   // [E][H][I]
    const int* __restrict__ counts,
    const int* __restrict__ tlist,
    const float* __restrict__ wlist,
    float* __restrict__ out) {
  __shared__ _Float16 xa[MT * XAS];   // 32 x 1024 x-tile (f16), padded
  __shared__ _Float16 al[MT * ALS];   // 32 x 128 activation tile (f16), padded
  __shared__ int   tokS[MT];
  __shared__ float wtS[MT];

  int e    = blockIdx.x / TILES_MAX;
  int tile = blockIdx.x % TILES_MAX;
  int cnt  = counts[e];
  if (tile * MT >= cnt) return;       // uniform early-exit

  int tid  = threadIdx.x;
  int lane = tid & 31;
  int w    = tid >> 5;                // wave id 0..7
  int lr   = lane & 15;
  int hi8  = (lane >> 4) * 8;

  if (tid < MT) {
    int s = tile * MT + tid;
    int tok = 0; float wt = 0.f;
    if (s < cnt) { tok = tlist[e * T_TOK + s]; wt = wlist[e * T_TOK + s]; }
    tokS[tid] = tok; wtS[tid] = wt;
  }
  __syncthreads();

  // stage x tile: 32 rows x 1024 f16, b128 chunks, coalesced
  for (int idx = tid; idx < MT * (Hd / 8); idx += 256) {
    int m  = idx >> 7;
    int kc = (idx & 127) << 3;
    u32x4 v = *(const u32x4*)(xh + (size_t)tokS[m] * Hd + kc);
    *(u32x4*)(xa + m * XAS + kc) = v;
  }
  __syncthreads();

  const _Float16* wgB = wg_h + (size_t)e * Id * Hd;
  const _Float16* wuB = wu_h + (size_t)e * Id * Hd;
  const _Float16* wdB = wd_h + (size_t)e * Hd * Id;

  v8f cacc[2][8];   // 32 x 128 fp32 output accumulator per wave
#pragma unroll
  for (int i = 0; i < 2; ++i)
#pragma unroll
    for (int j = 0; j < 8; ++j) cacc[i][j] = (v8f){};

  for (int ic = 0; ic < Id / 128; ++ic) {
    // ---- phase 1: g,u = x @ Wg/Wu for 16 I-columns per wave (K = 1024) ----
    v8f cg0 = {}, cg1 = {}, cu0 = {}, cu1 = {};
    int nb1 = ic * 128 + w * 16;
#pragma unroll 4
    for (int kk = 0; kk < Hd; kk += 32) {
      v16h a0 = ldsA(xa, XAS, 0,  kk, lane);
      v16h a1 = ldsA(xa, XAS, 16, kk, lane);
      v16h bg = globB(wgB, Hd, nb1, kk, lane);
      v16h bu = globB(wuB, Hd, nb1, kk, lane);
      cg0 = WMMA_F16(a0, bg, cg0);
      cg1 = WMMA_F16(a1, bg, cg1);
      cu0 = WMMA_F16(a0, bu, cu0);
      cu1 = WMMA_F16(a1, bu, cu1);
    }
    // silu(g) * u -> LDS (f16)
    {
      int ncol = w * 16 + lr;
#pragma unroll
      for (int r = 0; r < 8; ++r) {
        float g0 = cg0[r], u0 = cu0[r];
        float g1 = cg1[r], u1 = cu1[r];
        float a0v = u0 * g0 / (1.f + __expf(-g0));
        float a1v = u1 * g1 / (1.f + __expf(-g1));
        al[(hi8 + r) * ALS + ncol]        = (_Float16)a0v;
        al[(16 + hi8 + r) * ALS + ncol]   = (_Float16)a1v;
      }
    }
    __syncthreads();

    // ---- phase 2: Y += a(32x128) @ Wd(128 x 128-cols-of-this-wave) ----
#pragma unroll
    for (int ks = 0; ks < 4; ++ks) {
      v16h a0 = ldsA(al, ALS, 0,  ks * 32, lane);
      v16h a1 = ldsA(al, ALS, 16, ks * 32, lane);
#pragma unroll
      for (int nb = 0; nb < 8; ++nb) {
        v16h b = globB(wdB, Id, w * 128 + nb * 16, ic * 128 + ks * 32, lane);
        cacc[0][nb] = WMMA_F16(a0, b, cacc[0][nb]);
        cacc[1][nb] = WMMA_F16(a1, b, cacc[1][nb]);
      }
    }
    __syncthreads();   // al reused next chunk
  }

  // scale by routing weight, combine into out
#pragma unroll
  for (int mb = 0; mb < 2; ++mb)
#pragma unroll
    for (int nb = 0; nb < 8; ++nb) {
      v8f cv = cacc[mb][nb];
      int ncol = w * 128 + nb * 16 + lr;
#pragma unroll
      for (int r = 0; r < 8; ++r) {
        int m = mb * 16 + hi8 + r;
        float val = cv[r] * wtS[m];
        atomicAdd(out + (size_t)tokS[m] * Hd + ncol, val);
      }
    }
}

// ---------------- host launcher ----------------
extern "C" void kernel_launch(void* const* d_in, const int* in_sizes, int n_in,
                              void* d_out, int out_size, void* d_ws, size_t ws_size,
                              hipStream_t stream) {
  const float* x   = (const float*)d_in[0];   // [B,S,H]
  const float* wg  = (const float*)d_in[1];   // [H,E]
  const float* wgp = (const float*)d_in[2];   // [E,H,I]
  const float* wup = (const float*)d_in[3];   // [E,H,I]
  const float* wdp = (const float*)d_in[4];   // [E,I,H]
  float* out = (float*)d_out;

  char* ws = (char*)d_ws;
  size_t o = 0;
  _Float16* xh  = (_Float16*)(ws + o); o += (size_t)T_TOK * Hd * 2;
  _Float16* wgh = (_Float16*)(ws + o); o += (size_t)NE * Hd * Id * 2;
  _Float16* wuh = (_Float16*)(ws + o); o += (size_t)NE * Hd * Id * 2;
  _Float16* wdh = (_Float16*)(ws + o); o += (size_t)NE * Hd * Id * 2;
  int*   counts = (int*)(ws + o);      o += 256;
  int*   tlist  = (int*)(ws + o);      o += (size_t)NE * T_TOK * 4;
  float* wlist  = (float*)(ws + o);    o += (size_t)NE * T_TOK * 4;
  (void)ws_size; (void)in_sizes; (void)n_in;

  hipMemsetAsync(counts, 0, NE * sizeof(int), stream);
  hipMemsetAsync(d_out, 0, (size_t)out_size * sizeof(float), stream);

  router_kernel<<<T_TOK / 8, 256, 0, stream>>>(x, wg, counts, tlist, wlist);

  long n8 = (long)T_TOK * Hd / 8;
  cvt_x_kernel<<<(unsigned)((n8 + 255) / 256), 256, 0, stream>>>(x, xh, n8);

  dim3 tb(32, 8);
  transpose_cvt_kernel<<<dim3(Id / 32, Hd / 32, NE), tb, 0, stream>>>(wgp, wgh, Hd, Id);
  transpose_cvt_kernel<<<dim3(Id / 32, Hd / 32, NE), tb, 0, stream>>>(wup, wuh, Hd, Id);
  transpose_cvt_kernel<<<dim3(Hd / 32, Id / 32, NE), tb, 0, stream>>>(wdp, wdh, Id, Hd);

  moe_ffn_kernel<<<NE * TILES_MAX, 256, 0, stream>>>(
      xh, wgh, wuh, wdh, counts, tlist, wlist, out);
}
